// minivnGPT_52398601011747
// MI455X (gfx1250) — matrix operand order repaired and checked by reading, not compiled, not run on hardware
//
#include <hip/hip_runtime.h>
#include <hip/hip_bf16.h>
#include <math.h>

// ---------------------------------------------------------------------------
// mini-GPT forward for MI455X (gfx1250, wave32, WMMA).
// All matmuls: v_wmma_f32_16x16x32_bf16 (bf16 A/B, f32 accum).
// Attention: fused flash-style kernel (online softmax, causal) in LDS.
// A-tile / Q / K staging uses GLOBAL_LOAD_ASYNC_TO_LDS_B128 when available.
// ---------------------------------------------------------------------------

#define TT   2048
#define DD   1024
#define HH   16
#define DK   64
#define LL   4
#define VV   32000

typedef __bf16 bf16;
typedef __attribute__((ext_vector_type(16))) __bf16 v16bf;
typedef __attribute__((ext_vector_type(8)))  float  v8f;
typedef __attribute__((ext_vector_type(4)))  int    v4i;

#define AS_GLOBAL __attribute__((address_space(1)))
#define AS_LDS    __attribute__((address_space(3)))

// fp32 -> bf16 round-to-nearest-even
static __device__ __forceinline__ bf16 f2bf(float f) {
    unsigned u = __builtin_bit_cast(unsigned, f);
    unsigned r = u + 0x7FFFu + ((u >> 16) & 1u);
    unsigned short hs = (unsigned short)(r >> 16);
    return __builtin_bit_cast(bf16, hs);
}

struct bf16x16_bits { uint4 a, b; };

// ---- async global->LDS copy (16B per lane), ASYNCcnt-tracked --------------
#if defined(__has_builtin)
#if __has_builtin(__builtin_amdgcn_global_load_async_to_lds_b128) && \
    __has_builtin(__builtin_amdgcn_s_wait_asynccnt)
#define USE_ASYNC_LDS 1
#endif
#endif

static __device__ __forceinline__ void async_cp16(void* lds_dst, const void* gsrc) {
#ifdef USE_ASYNC_LDS
    __builtin_amdgcn_global_load_async_to_lds_b128(
        (AS_GLOBAL v4i*)gsrc, (AS_LDS v4i*)lds_dst, 0, 0);
#else
    *(uint4*)lds_dst = *(const uint4*)gsrc;
#endif
}

static __device__ __forceinline__ void wait_async() {
#ifdef USE_ASYNC_LDS
    __builtin_amdgcn_s_wait_asynccnt(0);
#endif
}

// A-fragment (16x32 bf16, M x K), per CDNA5 ISA "16-bit A-Matrix 16x32":
// lane l, group g=l/16, m=l%16; halves 0..7 -> K = k0+8g .. ; halves 8..15 -> K = k0+16+8g ..
static __device__ __forceinline__ v16bf load_a_frag(const bf16* lds, int stride,
                                                    int m0, int k0, int lane) {
    int m = m0 + (lane & 15), g = lane >> 4;
    const bf16* p = lds + m * stride + k0 + 8 * g;
    bf16x16_bits s;
    s.a = *(const uint4*)(p);
    s.b = *(const uint4*)(p + 16);
    return __builtin_bit_cast(v16bf, s);
}

// B-fragment (32x16 bf16, K x N), from a transposed (N x K) LDS tile:
// lane l: n = n0 + l%16, g=l/16; halves 0..7 -> K = k0+8g ..; halves 8..15 -> K = k0+16+8g ..
static __device__ __forceinline__ v16bf load_b_frag(const bf16* ldsT, int stride,
                                                    int n0, int k0, int lane) {
    int n = n0 + (lane & 15), g = lane >> 4;
    const bf16* p = ldsT + n * stride + k0 + 8 * g;
    bf16x16_bits s;
    s.a = *(const uint4*)(p);
    s.b = *(const uint4*)(p + 16);
    return __builtin_bit_cast(v16bf, s);
}

static __device__ __forceinline__ v8f wmma_bf16(v16bf a, v16bf b, v8f c) {
    return __builtin_amdgcn_wmma_f32_16x16x32_bf16(false, a, false, b,
                                                   (short)0, c, false, false);
}

// ---------------------------------------------------------------------------
// 1) Embedding + sinusoidal positional encoding -> h (f32, T x D)
// ---------------------------------------------------------------------------
__global__ void embed_pe_k(const int* __restrict__ x, const float* __restrict__ emb,
                           float* __restrict__ h) {
    int t = blockIdx.x, tid = threadIdx.x;
    int tok = x[t];
    for (int kk = 0; kk < 4; ++kk) {
        int d = tid + 256 * kk;
        float i = (float)(d >> 1);
        float ang = (float)t * powf(10000.0f, -2.0f * i / (float)DD);
        float pe = ((d & 1) == 0) ? sinf(ang) : cosf(ang);
        h[(size_t)t * DD + d] = emb[(size_t)tok * DD + d] + pe;
    }
}

// ---------------------------------------------------------------------------
// 2) LayerNorm row kernel: f32 in -> bf16 out (feeds WMMA A operands)
// ---------------------------------------------------------------------------
__global__ void layernorm_k(const float* __restrict__ x, const float* __restrict__ g,
                            const float* __restrict__ b, bf16* __restrict__ out) {
    int row = blockIdx.x, tid = threadIdx.x;
    const float* xr = x + (size_t)row * DD;
    float vals[4], s = 0.f, s2 = 0.f;
    for (int kk = 0; kk < 4; ++kk) {
        float v = xr[tid + 256 * kk];
        vals[kk] = v; s += v; s2 += v * v;
    }
    __shared__ float red[256], red2[256];
    red[tid] = s; red2[tid] = s2;
    __syncthreads();
    for (int off = 128; off > 0; off >>= 1) {
        if (tid < off) { red[tid] += red[tid + off]; red2[tid] += red2[tid + off]; }
        __syncthreads();
    }
    float mu   = red[0] * (1.0f / DD);
    float var  = red2[0] * (1.0f / DD) - mu * mu;
    float rstd = rsqrtf(var + 1e-5f);
    for (int kk = 0; kk < 4; ++kk) {
        int d = tid + 256 * kk;
        out[(size_t)row * DD + d] = f2bf((vals[kk] - mu) * rstd * g[d] + b[d]);
    }
}

// ---------------------------------------------------------------------------
// 3) WMMA GEMM: C[M,N] = A_bf16[M,K] * B_f32[K,N] (+bias)(+residual)
//    128x64 tile / 256-thread WG (8 wave32s). Each wave owns a 16-row strip
//    and all 4 column tiles: 1 A-frag reused across 4 WMMAs per k-step.
//    A tile staged via async global->LDS; B read coalesced fp32, converted +
//    transposed into LDS as bf16. blockIdx.z selects per-head slices (QKV).
// ---------------------------------------------------------------------------
template <int OUT_BF16, int RESID, int BIAS>
__global__ void gemm_k(const bf16* __restrict__ A, int lda,
                       const float* __restrict__ B, int ldb,
                       void* __restrict__ Cv, int ldc,
                       const float* __restrict__ bias,
                       const float* __restrict__ resid,
                       int Kd, size_t strideBz, size_t strideCz) {
    const int m0 = blockIdx.x * 128, n0 = blockIdx.y * 64, z = blockIdx.z;
    B += (size_t)z * strideBz;
    const int tid = threadIdx.x, lane = tid & 31, wave = tid >> 5;

    __shared__ bf16 At [128 * 40];      // 128 x 32 A tile (row stride 40 halves)
    __shared__ bf16 BTt[ 64 * 40];      // 64(n) x 32(k) transposed B tile

    v8f acc[4] = {{}, {}, {}, {}};
    for (int kk = 0; kk < Kd; kk += 32) {
        {   // stage A: 32B per thread via two async b128 copies
            int r = tid >> 1, cc = (tid & 1) * 16;
            const bf16* gp = A + (size_t)(m0 + r) * lda + kk + cc;
            async_cp16(&At[r * 40 + cc],     gp);
            async_cp16(&At[r * 40 + cc + 8], gp + 8);
        }
        {   // stage B: read 8 fp32 coalesced, convert+transpose into LDS
            int r = tid >> 3, cc = (tid & 7) * 8;
            const float* gp = B + (size_t)(kk + r) * ldb + n0 + cc;
            float4 f0 = *(const float4*)gp;
            float4 f1 = *(const float4*)(gp + 4);
            bf16 tmp[8] = { f2bf(f0.x), f2bf(f0.y), f2bf(f0.z), f2bf(f0.w),
                            f2bf(f1.x), f2bf(f1.y), f2bf(f1.z), f2bf(f1.w) };
            for (int e = 0; e < 8; ++e) BTt[(cc + e) * 40 + r] = tmp[e];
        }
        wait_async();
        __syncthreads();
        v16bf a = load_a_frag(At, 40, 16 * wave, 0, lane);
        for (int j = 0; j < 4; ++j) {
            v16bf b = load_b_frag(BTt, 40, 16 * j, 0, lane);
            acc[j] = wmma_bf16(a, b, acc[j]);
        }
        __syncthreads();
    }
    // Epilogue. C/D layout: VGPR r -> M = 16*wave + r + 8*(lane/16), N = 16*j + lane%16
    int roff = 8 * (lane >> 4), col = lane & 15;
    for (int j = 0; j < 4; ++j) {
        for (int r = 0; r < 8; ++r) {
            int row = m0 + 16 * wave + r + roff;
            int cN  = n0 + 16 * j + col;
            float v = acc[j][r];
            if (BIAS)  v += bias[cN];
            if (RESID) v += resid[(size_t)row * ldc + cN];
            if (OUT_BF16) {
                bf16* C = (bf16*)Cv + (size_t)z * strideCz;
                C[(size_t)row * ldc + cN] = f2bf(v);
            } else {
                float* C = (float*)Cv;
                C[(size_t)row * ldc + cN] = v;
            }
        }
    }
}

// ---------------------------------------------------------------------------
// 4) Fused causal flash attention. q/k/v: [H][T][DK] bf16, out attb: [T][D] bf16
//    One WG per (64-query block, head). S and P tiles live entirely in LDS.
// ---------------------------------------------------------------------------
__global__ void flash_attn_k(const bf16* __restrict__ q, const bf16* __restrict__ kmat,
                             const bf16* __restrict__ vmat, bf16* __restrict__ attb) {
    const int qb0  = blockIdx.x * 64;
    const int head = blockIdx.y;
    const int tid  = threadIdx.x, lane = tid & 31, wave = tid >> 5;
    const int ti = wave >> 1, tj = (wave & 1) * 2;
    const float scale = 0.125f;  // DK^-0.5

    __shared__ bf16  Qt [64 * 72];      // Q tile   (64 q x 64 dk)
    __shared__ bf16  KVt[64 * 72];      // K tile (keys x dk) then V^T (dk x keys)
    __shared__ float Sb [64 * 65];      // f32 scores
    __shared__ bf16  Pt [64 * 72];      // softmax probs
    __shared__ float mrow[64], lrow[64], crow[64];

    const bf16* qh = q + ((size_t)head * TT + qb0) * DK;
    {   int r = tid >> 2, c = (tid & 3) * 16;
        async_cp16(&Qt[r * 72 + c],     qh + r * DK + c);
        async_cp16(&Qt[r * 72 + c + 8], qh + r * DK + c + 8);
    }
    if (tid < 64) { mrow[tid] = -1e30f; lrow[tid] = 0.f; }
    v8f o0 = {}, o1 = {};
    wait_async();
    __syncthreads();

    for (int kb0 = 0; kb0 <= qb0; kb0 += 64) {
        const bf16* kh = kmat + ((size_t)head * TT + kb0) * DK;
        {   int r = tid >> 2, c = (tid & 3) * 16;
            async_cp16(&KVt[r * 72 + c],     kh + r * DK + c);
            async_cp16(&KVt[r * 72 + c + 8], kh + r * DK + c + 8);
        }
        wait_async();
        __syncthreads();
        // S = Q * K^T  (B operand = K tile rows: already N(=key) x K(=dk) layout)
        v8f s0 = {}, s1 = {};
        for (int ks = 0; ks < DK; ks += 32) {
            v16bf a  = load_a_frag(Qt, 72, 16 * ti, ks, lane);
            v16bf b0 = load_b_frag(KVt, 72, 16 * tj, ks, lane);
            v16bf b1 = load_b_frag(KVt, 72, 16 * (tj + 1), ks, lane);
            s0 = wmma_bf16(a, b0, s0);
            s1 = wmma_bf16(a, b1, s1);
        }
        {   int roff = 8 * (lane >> 4), col = lane & 15;
            for (int r = 0; r < 8; ++r) {
                Sb[(16 * ti + r + roff) * 65 + 16 * tj + col]       = s0[r];
                Sb[(16 * ti + r + roff) * 65 + 16 * (tj + 1) + col] = s1[r];
            }
        }
        __syncthreads();
        // Online softmax (one thread per query row)
        if (tid < 64) {
            int r = tid, qrow = qb0 + r;
            float bmax = -1e30f;
            for (int c = 0; c < 64; ++c) {
                float v = (kb0 + c <= qrow) ? Sb[r * 65 + c] * scale : -1e30f;
                bmax = fmaxf(bmax, v);
            }
            float mnew = fmaxf(mrow[r], bmax);
            float cc   = __expf(mrow[r] - mnew);
            float rs   = 0.f;
            for (int c = 0; c < 64; ++c) {
                float e = 0.f;
                if (kb0 + c <= qrow) e = __expf(Sb[r * 65 + c] * scale - mnew);
                Pt[r * 72 + c] = f2bf(e);
                rs += e;
            }
            lrow[r] = lrow[r] * cc + rs;
            crow[r] = cc;
            mrow[r] = mnew;
        }
        __syncthreads();
        // Rescale running O by correction factor
        {   int roff = 8 * (lane >> 4);
            for (int r = 0; r < 8; ++r) {
                float cc = crow[16 * ti + r + roff];
                o0[r] *= cc; o1[r] *= cc;
            }
        }
        // Restage KVt with V^T (dk-major) for the P*V WMMA B operand
        {   const bf16* vh = vmat + ((size_t)head * TT + kb0) * DK;
            int r = tid >> 2, c = (tid & 3) * 16;
            bf16 tmp[16];
            *(uint4*)&tmp[0] = *(const uint4*)(vh + r * DK + c);
            *(uint4*)&tmp[8] = *(const uint4*)(vh + r * DK + c + 8);
            for (int e = 0; e < 16; ++e) KVt[(c + e) * 72 + r] = tmp[e];
        }
        __syncthreads();
        // O += P * V
        for (int ks = 0; ks < 64; ks += 32) {
            v16bf a  = load_a_frag(Pt, 72, 16 * ti, ks, lane);
            v16bf b0 = load_b_frag(KVt, 72, 16 * tj, ks, lane);
            v16bf b1 = load_b_frag(KVt, 72, 16 * (tj + 1), ks, lane);
            o0 = wmma_bf16(a, b0, o0);
            o1 = wmma_bf16(a, b1, o1);
        }
        __syncthreads();
    }
    // Epilogue: normalize, write [t][head*64 + dk] bf16 (head-concat layout)
    {   int roff = 8 * (lane >> 4), col = lane & 15;
        for (int r = 0; r < 8; ++r) {
            int row = 16 * ti + r + roff;
            float inv = 1.0f / lrow[row];
            size_t o = (size_t)(qb0 + row) * DD + head * DK;
            attb[o + 16 * tj + col]       = f2bf(o0[r] * inv);
            attb[o + 16 * (tj + 1) + col] = f2bf(o1[r] * inv);
        }
    }
}

// ---------------------------------------------------------------------------
// Host launcher. Workspace (~32 MB): h(f32) + yb/qb/kb/vb/attb/hb (bf16).
// ---------------------------------------------------------------------------
extern "C" void kernel_launch(void* const* d_in, const int* in_sizes, int n_in,
                              void* d_out, int out_size, void* d_ws, size_t ws_size,
                              hipStream_t stream) {
    (void)in_sizes; (void)n_in; (void)out_size; (void)ws_size;
    const int*   x     = (const int*)  d_in[0];
    const float* emb   = (const float*)d_in[1];
    const float* Wq    = (const float*)d_in[2];
    const float* Wk    = (const float*)d_in[3];
    const float* Wv    = (const float*)d_in[4];
    const float* Wproj = (const float*)d_in[5];
    const float* bproj = (const float*)d_in[6];
    const float* ln1_g = (const float*)d_in[7];
    const float* ln1_b = (const float*)d_in[8];
    const float* Wffn  = (const float*)d_in[9];
    const float* bffn  = (const float*)d_in[10];
    const float* ln2_g = (const float*)d_in[11];
    const float* ln2_b = (const float*)d_in[12];
    const float* lnf_g = (const float*)d_in[13];
    const float* lnf_b = (const float*)d_in[14];
    const float* Wout  = (const float*)d_in[15];
    const float* bout  = (const float*)d_in[16];
    float* out = (float*)d_out;

    char* w = (char*)d_ws;
    float* h   = (float*)w;  w += (size_t)TT * DD * sizeof(float);
    bf16* yb   = (bf16*)w;   w += (size_t)TT * DD * sizeof(bf16);
    bf16* qb   = (bf16*)w;   w += (size_t)TT * DD * sizeof(bf16);
    bf16* kb   = (bf16*)w;   w += (size_t)TT * DD * sizeof(bf16);
    bf16* vb   = (bf16*)w;   w += (size_t)TT * DD * sizeof(bf16);
    bf16* attb = (bf16*)w;   w += (size_t)TT * DD * sizeof(bf16);
    bf16* hb   = (bf16*)w;   w += (size_t)TT * DD * sizeof(bf16);

    embed_pe_k<<<TT, 256, 0, stream>>>(x, emb, h);

    for (int l = 0; l < LL; ++l) {
        layernorm_k<<<TT, 256, 0, stream>>>(h, ln1_g + l * DD, ln1_b + l * DD, yb);

        // Q/K/V: per head (blockIdx.z), B = W[l][h] (D x DK), out bf16 [H][T][DK]
        dim3 gqkv(TT / 128, 1, HH);
        size_t sBz = (size_t)DD * DK, sCz = (size_t)TT * DK;
        gemm_k<1, 0, 0><<<gqkv, 256, 0, stream>>>(yb, DD, Wq + (size_t)l * HH * DD * DK,
                                                  DK, qb, DK, nullptr, nullptr, DD, sBz, sCz);
        gemm_k<1, 0, 0><<<gqkv, 256, 0, stream>>>(yb, DD, Wk + (size_t)l * HH * DD * DK,
                                                  DK, kb, DK, nullptr, nullptr, DD, sBz, sCz);
        gemm_k<1, 0, 0><<<gqkv, 256, 0, stream>>>(yb, DD, Wv + (size_t)l * HH * DD * DK,
                                                  DK, vb, DK, nullptr, nullptr, DD, sBz, sCz);

        flash_attn_k<<<dim3(TT / 64, HH), 256, 0, stream>>>(qb, kb, vb, attb);

        // h = h + attb @ Wproj + bproj
        gemm_k<0, 1, 1><<<dim3(TT / 128, DD / 64), 256, 0, stream>>>(
            attb, DD, Wproj + (size_t)l * DD * DD, DD, h, DD,
            bproj + l * DD, h, DD, 0, 0);

        layernorm_k<<<TT, 256, 0, stream>>>(h, ln2_g + l * DD, ln2_b + l * DD, yb);

        // h = h + yb @ Wffn + bffn
        gemm_k<0, 1, 1><<<dim3(TT / 128, DD / 64), 256, 0, stream>>>(
            yb, DD, Wffn + (size_t)l * DD * DD, DD, h, DD,
            bffn + l * DD, h, DD, 0, 0);
    }

    layernorm_k<<<TT, 256, 0, stream>>>(h, lnf_g, lnf_b, hb);

    // logits = hb @ Wout + bout  (2048 x 32000 x 1024, the dominant GEMM)
    gemm_k<0, 0, 1><<<dim3(TT / 128, VV / 64), 256, 0, stream>>>(
        hb, DD, Wout, VV, out, VV, bout, nullptr, DD, 0, 0);
}
